// Mask_Attention_6158983102793
// MI455X (gfx1250) — compile-verified, matrix operand order checked by reading, not run on hardware
//
#include <hip/hip_runtime.h>
#include <hip/hip_bf16.h>

// ---------------------------------------------------------------------------
// Masked attention for MI455X (gfx1250, wave32, WMMA + async LDS copy).
//
//   S = C·Qᵀ ; mask(S==0 -> -inf) ; P = softmax_q(S) ; O = P·Q
//   B=8, LQ=LC=2048, D=256  (fp32 in / fp32 out)
//
// Compute-bound (~690 FLOP/byte) => both GEMMs via v_wmma_f32_16x16x32_bf16.
// Per block, the 32-query chunk (Q 32x256 + Qᵀ 256x32 bf16, 32 KB) is shared
// by all 8 waves, so it is staged into LDS once with
// global_load_async_to_lds_b128 (ASYNCcnt), double-buffered so chunk i+1
// streams in while chunk i is computed from LDS (ds_load_b128 fragments,
// padded strides => bank-conflict-free). Sᵀ/Oᵀ orientation keeps all
// fragment reads row-major and lets Pᵀ accumulators feed GEMM2's B operand
// with a pure per-lane f32->bf16 pack.
// ---------------------------------------------------------------------------

typedef __bf16 bf16_t;
typedef __attribute__((ext_vector_type(16))) __bf16 v16bf;
typedef __attribute__((ext_vector_type(8)))  __bf16 v8bf;
typedef __attribute__((ext_vector_type(4)))  __bf16 v4bf;
typedef __attribute__((ext_vector_type(8)))  float  v8f;

#define B_   8
#define LQ_  2048
#define LC_  2048
#define D_   256

#define QT_PAD   264   // 32-row Q tile row stride (elems): 528B = 132 dw -> banks 4r
#define QTT_PAD  40    // 256-row Qᵀ tile row stride (elems): 80B = 20 dw -> conflict-free

// Build one 16x32 bf16 WMMA A/B fragment from two contiguous 16B chunks.
// 16-bit operand layout (ISA 7.12.2): lanes 0-15 hold row (lane&15),
// K = {0..7, 16..23}; lanes 16-31 same row, K = {8..15, 24..31}.
__device__ inline v16bf frag_from_ptr(const bf16_t* p) {
    v8bf lo = *(const v8bf*)(p);
    v8bf hi = *(const v8bf*)(p + 16);
    return __builtin_shufflevector(lo, hi,
        0, 1, 2, 3, 4, 5, 6, 7, 8, 9, 10, 11, 12, 13, 14, 15);
}

__device__ inline v16bf load_frag_global(const bf16_t* __restrict__ base,
                                         int row, int stride, int col) {
    const int lane = threadIdx.x & 31;
    const int r = row + (lane & 15);
    const int c = col + ((lane & 16) ? 8 : 0);
    return frag_from_ptr(base + (size_t)r * stride + c);
}

// fp32 -> bf16 elementwise convert (n divisible by 4)
__global__ __launch_bounds__(256)
void cvt_bf16_kernel(const float* __restrict__ src, bf16_t* __restrict__ dst,
                     unsigned n) {
    unsigned i = (blockIdx.x * 256u + threadIdx.x) * 4u;
    if (i < n) {
        float4 v = *(const float4*)(src + i);
        v4bf o;
        o[0] = (bf16_t)v.x; o[1] = (bf16_t)v.y;
        o[2] = (bf16_t)v.z; o[3] = (bf16_t)v.w;
        *(v4bf*)(dst + i) = o;
    }
}

// [B, LQ, D] fp32 -> [B, D, LQ] bf16, LDS-tiled 32x32 transpose
__global__ __launch_bounds__(256)
void transpose_bf16_kernel(const float* __restrict__ src,
                           bf16_t* __restrict__ dst) {
    __shared__ bf16_t tile[32][33];
    const int b  = blockIdx.z;
    const int q0 = blockIdx.x * 32;
    const int d0 = blockIdx.y * 32;
    const int tx = threadIdx.x & 31;
    const int ty = threadIdx.x >> 5;
#pragma unroll
    for (int i = 0; i < 32; i += 8)
        tile[ty + i][tx] =
            (bf16_t)src[((size_t)b * LQ_ + q0 + ty + i) * D_ + d0 + tx];
    __syncthreads();
#pragma unroll
    for (int i = 0; i < 32; i += 8)
        dst[((size_t)b * D_ + d0 + ty + i) * LQ_ + q0 + tx] = tile[tx][ty + i];
}

// One wave owns 16 context rows; block (8 waves) owns 128.
// Per 32-query chunk: 16 WMMA for Sᵀ, 16 WMMA for Oᵀ.
__global__ __launch_bounds__(256)
void attn_kernel(const bf16_t* __restrict__ Qbf,   // [B, LQ, D]
                 const bf16_t* __restrict__ QbfT,  // [B, D, LQ]
                 const bf16_t* __restrict__ Cbf,   // [B, LC, D]
                 float* __restrict__ out) {        // [B, LC, D]
    __shared__ __align__(16) bf16_t qtile [2][32][QT_PAD];   // Q chunk  (q x d)
    __shared__ __align__(16) bf16_t qttile[2][256][QTT_PAD]; // Qᵀ chunk (d x q)

    const int tid   = threadIdx.x;
    const int wave  = tid >> 5;
    const int lane  = tid & 31;
    const int b     = blockIdx.y;
    const int cbase = blockIdx.x * 128 + wave * 16;

    const bf16_t* Qb  = Qbf  + (size_t)b * LQ_ * D_;
    const bf16_t* QTb = QbfT + (size_t)b * D_ * LQ_;
    const bf16_t* Cb  = Cbf  + (size_t)b * LC_ * D_;

    // Issue the async global->LDS copy of one 32-query chunk (all 256 thr).
    auto issue_chunk = [&](int q0, int buf) {
        // Q tile: 32 rows x 256 cols; 8 threads/row, 4 x 16B segments each.
        {
            const int r  = tid >> 3;
            const int s0 = (tid & 7) * 4;               // segment index (16B units)
            const bf16_t* gp = Qb + (size_t)(q0 + r) * D_ + s0 * 8;
            unsigned lp = (unsigned)(uintptr_t)&qtile[buf][r][s0 * 8];
#pragma unroll
            for (int j = 0; j < 4; ++j)
                asm volatile("global_load_async_to_lds_b128 %0, %1, off"
                             :: "v"(lp + j * 16), "v"(gp + j * 8) : "memory");
        }
        // Qᵀ tile: 256 rows x 32 cols; 1 thread/row, 4 x 16B segments.
        {
            const bf16_t* gp = QTb + (size_t)tid * LQ_ + q0;
            unsigned lp = (unsigned)(uintptr_t)&qttile[buf][tid][0];
#pragma unroll
            for (int j = 0; j < 4; ++j)
                asm volatile("global_load_async_to_lds_b128 %0, %1, off"
                             :: "v"(lp + j * 16), "v"(gp + j * 8) : "memory");
        }
    };

    // Resident context fragments: B operand of GEMM1 (K=d x N=c), lane = c.
    v16bf cfrag[8];
#pragma unroll
    for (int k = 0; k < 8; ++k)
        cfrag[k] = load_frag_global(Cb, cbase, D_, 32 * k);

    // Oᵀ accumulators: 16 tiles of 16(d) x 16(c), lane = c.
    v8f oacc[16];
#pragma unroll
    for (int i = 0; i < 16; ++i)
        oacc[i] = (v8f){0.f,0.f,0.f,0.f,0.f,0.f,0.f,0.f};

    float m_run = -INFINITY;
    float l_run = 0.f;
    const float LOG2E = 1.44269504088896341f;

    const int rsel = lane & 15;
    const int csel = (lane & 16) ? 8 : 0;

    issue_chunk(0, 0);                       // prologue

    const int NCHUNK = LQ_ / 32;
    for (int it = 0; it < NCHUNK; ++it) {
        const int buf = it & 1;
        const int q0  = it * 32;

        asm volatile("s_wait_asynccnt 0x0" ::: "memory"); // own chunk-i arrived
        __syncthreads();                                  // everyone's arrived;
                                                          // also: all waves done
                                                          // reading buf^1 (i-1)
        if (it + 1 < NCHUNK)
            issue_chunk(q0 + 32, buf ^ 1);   // overlaps with compute below

        // --- GEMM1: Sᵀ = Q_chunk · Cᵀ (two 16q x 16c tiles), A from LDS ---
        v8f s[2];
#pragma unroll
        for (int t = 0; t < 2; ++t) {
            v8f acc = (v8f){0.f,0.f,0.f,0.f,0.f,0.f,0.f,0.f};
#pragma unroll
            for (int k = 0; k < 8; ++k) {
                v16bf a = frag_from_ptr(&qtile[buf][16 * t + rsel][32 * k + csel]);
                acc = __builtin_amdgcn_wmma_f32_16x16x32_bf16(
                    false, a, false, cfrag[k], (short)0, acc, false, false);
            }
            s[t] = acc;
        }

        // --- mask (==0 -> -inf) + chunk max per lane (= per c) ---
        float cmax = -INFINITY;
#pragma unroll
        for (int t = 0; t < 2; ++t)
#pragma unroll
            for (int r = 0; r < 8; ++r) {
                float v = s[t][r];
                v = (v == 0.f) ? -INFINITY : v;
                s[t][r] = v;
                cmax = fmaxf(cmax, v);
            }
        cmax = fmaxf(cmax, __shfl_xor(cmax, 16, 32));

        const float m_new = fmaxf(m_run, cmax);
        float alpha, lsum = 0.f;
        if (m_new == -INFINITY) {            // fully-masked so far
            alpha = 1.f;
#pragma unroll
            for (int t = 0; t < 2; ++t)
#pragma unroll
                for (int r = 0; r < 8; ++r) s[t][r] = 0.f;
        } else {
            alpha = __builtin_amdgcn_exp2f((m_run - m_new) * LOG2E);
#pragma unroll
            for (int t = 0; t < 2; ++t)
#pragma unroll
                for (int r = 0; r < 8; ++r) {
                    float p = __builtin_amdgcn_exp2f((s[t][r] - m_new) * LOG2E);
                    s[t][r] = p;
                    lsum += p;
                }
        }
        lsum += __shfl_xor(lsum, 16, 32);
        l_run = l_run * alpha + lsum;
        m_run = m_new;

        // --- rescale Oᵀ accumulators (per-lane factor = per-c) ---
#pragma unroll
        for (int i = 0; i < 16; ++i)
#pragma unroll
            for (int r = 0; r < 8; ++r) oacc[i][r] *= alpha;

        // --- pack Pᵀ tiles into GEMM2 B operand (32q x 16c): per-lane only ---
        v16bf pfrag;
#pragma unroll
        for (int j = 0; j < 8; ++j) {
            const int t  = j >> 2;
            const int rr = (j & 3) * 2;
            pfrag[2 * j]     = (bf16_t)s[t][rr];
            pfrag[2 * j + 1] = (bf16_t)s[t][rr + 1];
        }

        // --- GEMM2: Oᵀ += Qᵀ_chunk · Pᵀ (16 d-tiles), A from LDS ---
#pragma unroll
        for (int i = 0; i < 16; ++i) {
            v16bf a = frag_from_ptr(&qttile[buf][16 * i + rsel][csel]);
            oacc[i] = __builtin_amdgcn_wmma_f32_16x16x32_bf16(
                false, a, false, pfrag, (short)0, oacc[i], false, false);
        }
    }

    // --- finalize: O = Oᵀ/l (0 if row fully masked), store row-major ---
    const float inv_l = (l_run > 0.f) ? (1.f / l_run) : 0.f;
    const int c = cbase + rsel;
    float* orow = out + ((size_t)b * LC_ + c) * D_;
#pragma unroll
    for (int i = 0; i < 16; ++i) {
        const int d0 = 16 * i + csel;
        float4 lo = make_float4(oacc[i][0] * inv_l, oacc[i][1] * inv_l,
                                oacc[i][2] * inv_l, oacc[i][3] * inv_l);
        float4 hi = make_float4(oacc[i][4] * inv_l, oacc[i][5] * inv_l,
                                oacc[i][6] * inv_l, oacc[i][7] * inv_l);
        *(float4*)(orow + d0)     = lo;
        *(float4*)(orow + d0 + 4) = hi;
    }
}

extern "C" void kernel_launch(void* const* d_in, const int* in_sizes, int n_in,
                              void* d_out, int out_size, void* d_ws,
                              size_t ws_size, hipStream_t stream) {
    (void)in_sizes; (void)n_in; (void)out_size; (void)ws_size;
    const float* q = (const float*)d_in[0];   // [B, LQ, D]
    const float* c = (const float*)d_in[1];   // [B, LC, D]
    float* out = (float*)d_out;               // [B, LC, D]

    const size_t nq = (size_t)B_ * LQ_ * D_;  // 4M elems
    const size_t nc = (size_t)B_ * LC_ * D_;  // 4M elems
    // ws layout: Qbf | QbfT | Cbf (bf16) => ~25 MiB
    bf16_t* Qbf  = (bf16_t*)d_ws;
    bf16_t* QbfT = Qbf + nq;
    bf16_t* Cbf  = QbfT + nq;

    cvt_bf16_kernel<<<(unsigned)(nq / 4 / 256), 256, 0, stream>>>(q, Qbf, (unsigned)nq);
    cvt_bf16_kernel<<<(unsigned)(nc / 4 / 256), 256, 0, stream>>>(c, Cbf, (unsigned)nc);

    dim3 tg(LQ_ / 32, D_ / 32, B_);
    transpose_bf16_kernel<<<tg, 256, 0, stream>>>(q, QbfT);

    dim3 grid(LC_ / 128, B_);
    attn_kernel<<<grid, 256, 0, stream>>>(Qbf, QbfT, Cbf, out);
}
